// VRWKV_SpatialMix_44160853737952
// MI455X (gfx1250) — compile-verified, matrix operand order checked by reading, not run on hardware
//
#include <hip/hip_runtime.h>
#include <hip/hip_bf16.h>

typedef __bf16 bf16_t;
typedef __attribute__((ext_vector_type(16))) __bf16 v16bf;
typedef __attribute__((ext_vector_type(8)))  __bf16 v8bf;
typedef __attribute__((ext_vector_type(8)))  float  v8f;

#define EPS_V   1e-6f
#define CLAMP_V 10000.0f

// Async global->LDS DMA path (ASYNCcnt-tracked). The ROCm builtin for the load
// uses HIP LangAS pointer types that cannot be produced portably from HIP
// source, so the load itself goes through inline asm; the wait uses the
// builtin (probe-confirmed present when this macro is enabled).
#if defined(__gfx1250__) && __has_builtin(__builtin_amdgcn_s_wait_asynccnt)
#define USE_ASYNC_LDS 1
#endif

__device__ __forceinline__ float sigmoidf_(float x) { return 1.0f / (1.0f + __expf(-x)); }

// 16-byte global -> LDS copy; async DMA when available.
__device__ __forceinline__ void copy16_g2l(const bf16_t* g, bf16_t* l) {
#ifdef USE_ASYNC_LDS
  unsigned lofs = (unsigned)(uintptr_t)l;  // low 32 bits of generic ptr == LDS byte address
  asm volatile("global_load_async_to_lds_b128 %0, %1, off"
               :: "v"(lofs), "v"(g)
               : "memory");
#else
  *(v8bf*)l = *(const v8bf*)g;
#endif
}

__device__ __forceinline__ void wait_async_copies() {
#ifdef USE_ASYNC_LDS
  __builtin_amdgcn_s_wait_asynccnt(0);
#endif
}

// ---------------- f32 -> bf16 conversion ----------------
__global__ void cvt_f32_bf16_kernel(const float* __restrict__ src, bf16_t* __restrict__ dst, int n) {
  int i = blockIdx.x * blockDim.x + threadIdx.x;
  if (i < n) dst[i] = (bf16_t)src[i];
}

// ---------------- q_shift + zigzag reorder, emit bf16 A matrix ----------------
// xs[b,t,c] = qshift(x)[b, c, order[t]] ; order[t] for zigzag computed analytically.
__global__ void qshift_kernel(const float* __restrict__ x, bf16_t* __restrict__ xsb,
                              int B, int T, int C, int h, int w) {
  int idx = blockIdx.x * blockDim.x + threadIdx.x;
  int total = B * T * C;
  if (idx >= total) return;
  int c  = idx % C;
  int tc = idx / C;
  int t  = tc % T;
  int b  = tc / T;
  int i  = t / w;
  int jj = t - i * w;
  int j  = (i & 1) ? (w - 1 - jj) : jj;     // zigzag source position
  int g  = C >> 2;                           // gamma = 0.25
  int grp = c / g;
  int si = i, sj = j;
  if      (grp == 0) sj = j - 1;
  else if (grp == 1) sj = j + 1;
  else if (grp == 2) si = i - 1;
  else               si = i + 1;
  float val = 0.0f;
  if (si >= 0 && si < h && sj >= 0 && sj < w)
    val = x[((size_t)b * T + (size_t)si * w + sj) * C + c];
  xsb[idx] = (bf16_t)val;
}

// ---------------- bf16 WMMA GEMM: Out[m,n] = sum_k A[m,k] * Bw[n,k] ----------------
// Double-buffered async global->LDS pipeline; epilogue: 0 = none, 1 = sigmoid.
#define BM 128
#define BN 128
#define BK 32

union V16U { v16bf v; v8bf h[2]; };

__global__ __launch_bounds__(256) void gemm_bf16_wmma_kernel(
    const bf16_t* __restrict__ A, const bf16_t* __restrict__ Bw,
    float* __restrict__ Out, int M, int N, int Kd, int epilogue) {
  __shared__ bf16_t As[2][BM][BK];   // 2 x 8 KB
  __shared__ bf16_t Bs[2][BN][BK];   // 2 x 8 KB

  int tid   = threadIdx.x;
  int lane  = tid & 31;           // wave32
  int wv    = tid >> 5;           // 8 waves
  int waveM = wv >> 1;            // 0..3 -> 32-row strip
  int waveN = wv & 1;             // 0..1 -> 64-col strip
  int bM = blockIdx.x * BM;
  int bN = blockIdx.y * BN;

  v8f acc[2][4] = {};

  int ldRow  = tid >> 1;          // 0..127
  int ldHalf = (tid & 1) * 16;    // element offset into 32-wide K row
  int mrow   = lane & 15;         // M (or N) index within 16x16 tile
  int khalf  = (lane >> 4) * 8;   // K sub-offset per ISA 16-bit A/B layout

  const bf16_t* gaBase = A  + (size_t)(bM + ldRow) * Kd + ldHalf;
  const bf16_t* gbBase = Bw + (size_t)(bN + ldRow) * Kd + ldHalf;

  // Stage one BM/BN x BK tile into LDS buffer `buf` from K-offset k0.
  auto stage = [&](int buf, int k0) {
    const bf16_t* ga = gaBase + k0;
    copy16_g2l(ga,     &As[buf][ldRow][ldHalf]);
    copy16_g2l(ga + 8, &As[buf][ldRow][ldHalf + 8]);
    const bf16_t* gb = gbBase + k0;
    copy16_g2l(gb,     &Bs[buf][ldRow][ldHalf]);
    copy16_g2l(gb + 8, &Bs[buf][ldRow][ldHalf + 8]);
  };

  const int nK = Kd / BK;
  stage(0, 0);
  wait_async_copies();
  __syncthreads();

  for (int kt = 0; kt < nK; ++kt) {
    int cur = kt & 1;
    int nxt = cur ^ 1;
    if (kt + 1 < nK) stage(nxt, (kt + 1) * BK);  // DMA next tile while computing

    // Fragment loads: lanes 0-15 -> K {0..7,16..23}, lanes 16-31 -> K {8..15,24..31}
    V16U afrag[2], bfrag[4];
#pragma unroll
    for (int m = 0; m < 2; ++m) {
      int ar = waveM * 32 + m * 16 + mrow;
      afrag[m].h[0] = *(const v8bf*)&As[cur][ar][khalf];
      afrag[m].h[1] = *(const v8bf*)&As[cur][ar][16 + khalf];
    }
#pragma unroll
    for (int n = 0; n < 4; ++n) {
      int br = waveN * 64 + n * 16 + mrow;
      bfrag[n].h[0] = *(const v8bf*)&Bs[cur][br][khalf];
      bfrag[n].h[1] = *(const v8bf*)&Bs[cur][br][16 + khalf];
    }
#pragma unroll
    for (int m = 0; m < 2; ++m)
#pragma unroll
      for (int n = 0; n < 4; ++n)
        acc[m][n] = __builtin_amdgcn_wmma_f32_16x16x32_bf16(
            false, afrag[m].v, false, bfrag[n].v, (short)0, acc[m][n], false, false);

    wait_async_copies();   // next tile resident in LDS
    __syncthreads();       // all waves done reading `cur`, writes to `nxt` visible
  }

  // Epilogue: D layout — VGPR r holds M = r (lanes 0-15) / M = 8+r (lanes 16-31), N = lane%16
#pragma unroll
  for (int m = 0; m < 2; ++m) {
    int rowBase = bM + waveM * 32 + m * 16 + (lane >> 4) * 8;
#pragma unroll
    for (int n = 0; n < 4; ++n) {
      int col = bN + waveN * 64 + n * 16 + (lane & 15);
#pragma unroll
      for (int r = 0; r < 8; ++r) {
        float vacc = acc[m][n][r];
        if (epilogue == 1) vacc = sigmoidf_(vacc);
        Out[(size_t)(rowBase + r) * N + col] = vacc;
      }
    }
  }
}

// ---------------- WKV double scan (one thread per (b,c) channel column) ----------------
// Pass 1: natural t order over (K,V) -> Y1. Pass 2: inputs permuted by zigzag order
// (K[order[t]], Y1[order[t]]) -> V2 (may alias V; each thread only touches its column).
__global__ void wkv_kernel(const float* __restrict__ Kp, const float* Vin,
                           const float* __restrict__ sdecay, const float* __restrict__ sfirst,
                           float* __restrict__ Y1, float* V2,
                           int B, int T, int C, int h, int w) {
  int idx = blockIdx.x * blockDim.x + threadIdx.x;
  if (idx >= B * C) return;
  int c = idx % C;
  int b = idx / C;
  float Tf = (float)T;
  const size_t base = (size_t)b * T * C + c;

  // pass 1
  {
    float wd = sigmoidf_(sdecay[c] / Tf);
    wd = fminf(fmaxf(wd, 1e-5f), 1.0f - 1e-5f);
    float u = tanhf(sfirst[c] / Tf);
    float num = 0.f, den = 0.f, st = 0.f;
    for (int t = 0; t < T; ++t) {
      size_t o = base + (size_t)t * C;
      float kt = Kp[o], vt = Vin[o];
      num = num * wd + kt * vt;
      den = den * wd + fabsf(kt);
      st = u * st + num / (den + EPS_V);
      st = fminf(fmaxf(st, -CLAMP_V), CLAMP_V);
      Y1[o] = st;
    }
  }
  // pass 2
  {
    float wd = sigmoidf_(sdecay[C + c] / Tf);
    wd = fminf(fmaxf(wd, 1e-5f), 1.0f - 1e-5f);
    float u = tanhf(sfirst[C + c] / Tf);
    float num = 0.f, den = 0.f, st = 0.f;
    for (int t = 0; t < T; ++t) {
      int i = t / w;
      int jj = t - i * w;
      int j = (i & 1) ? (w - 1 - jj) : jj;
      size_t op = base + ((size_t)i * w + j) * C;
      float kt = Kp[op], vt = Y1[op];
      num = num * wd + kt * vt;
      den = den * wd + fabsf(kt);
      st = u * st + num / (den + EPS_V);
      st = fminf(fmaxf(st, -CLAMP_V), CLAMP_V);
      V2[base + (size_t)t * C] = st;
    }
  }
}

// ---------------- sr * v2 -> bf16 for final GEMM ----------------
__global__ void rv_kernel(const float* __restrict__ R, const float* __restrict__ V2,
                          bf16_t* __restrict__ rvb, int n) {
  int i = blockIdx.x * blockDim.x + threadIdx.x;
  if (i < n) rvb[i] = (bf16_t)(R[i] * V2[i]);
}

// ---------------- launch ----------------
extern "C" void kernel_launch(void* const* d_in, const int* in_sizes, int n_in,
                              void* d_out, int out_size, void* d_ws, size_t ws_size,
                              hipStream_t stream) {
  const float* x   = (const float*)d_in[0];
  const float* wk  = (const float*)d_in[1];
  const float* wvl = (const float*)d_in[2];
  const float* wr  = (const float*)d_in[3];
  const float* wo  = (const float*)d_in[4];
  const float* sd  = (const float*)d_in[5];
  const float* sf  = (const float*)d_in[6];

  // C from w_key (C x C); h,w fixed by the harness setup (56x56)
  int C = 1;
  while ((long long)C * C < (long long)in_sizes[1]) ++C;
  const int h = 56, w = 56;
  const int T = h * w;
  const int B = in_sizes[0] / (T * C);
  const size_t BT = (size_t)B * T;
  const size_t NE = BT * (size_t)C;

  // carve workspace (256B aligned slabs)
  char* p = (char*)d_ws;
  auto alloc = [&](size_t bytes) -> void* {
    void* r = (void*)p;
    p += (bytes + 255) & ~(size_t)255;
    return r;
  };
  bf16_t* xsb = (bf16_t*)alloc(NE * 2);
  bf16_t* wbk = (bf16_t*)alloc((size_t)C * C * 2);
  bf16_t* wbv = (bf16_t*)alloc((size_t)C * C * 2);
  bf16_t* wbr = (bf16_t*)alloc((size_t)C * C * 2);
  bf16_t* wbo = (bf16_t*)alloc((size_t)C * C * 2);
  float*  Kb  = (float*)alloc(NE * 4);
  float*  Vb  = (float*)alloc(NE * 4);   // reused as V2 after pass 1 consumes it
  float*  Rb  = (float*)alloc(NE * 4);
  float*  Y1  = (float*)alloc(NE * 4);
  bf16_t* rvb = (bf16_t*)alloc(NE * 2);

  const int cc = C * C;
  cvt_f32_bf16_kernel<<<(cc + 255) / 256, 256, 0, stream>>>(wk,  wbk, cc);
  cvt_f32_bf16_kernel<<<(cc + 255) / 256, 256, 0, stream>>>(wvl, wbv, cc);
  cvt_f32_bf16_kernel<<<(cc + 255) / 256, 256, 0, stream>>>(wr,  wbr, cc);
  cvt_f32_bf16_kernel<<<(cc + 255) / 256, 256, 0, stream>>>(wo,  wbo, cc);

  const int tot = (int)NE;
  qshift_kernel<<<(tot + 255) / 256, 256, 0, stream>>>(x, xsb, B, T, C, h, w);

  dim3 grid((unsigned)(BT / BM), (unsigned)(C / BN));
  gemm_bf16_wmma_kernel<<<grid, 256, 0, stream>>>(xsb, wbk, Kb, (int)BT, C, C, 0);
  gemm_bf16_wmma_kernel<<<grid, 256, 0, stream>>>(xsb, wbv, Vb, (int)BT, C, C, 0);
  gemm_bf16_wmma_kernel<<<grid, 256, 0, stream>>>(xsb, wbr, Rb, (int)BT, C, C, 1);

  const int bc = B * C;
  wkv_kernel<<<(bc + 255) / 256, 256, 0, stream>>>(Kb, Vb, sd, sf, Y1, Vb, B, T, C, h, w);

  rv_kernel<<<(tot + 255) / 256, 256, 0, stream>>>(Rb, Vb, rvb, tot);

  gemm_bf16_wmma_kernel<<<grid, 256, 0, stream>>>(rvb, wbo, (float*)d_out, (int)BT, C, C, 0);
}
